// AttentionWithBias_53042846106120
// MI455X (gfx1250) — matrix operand order repaired and checked by reading, not compiled, not run on hardware
//
#include <hip/hip_runtime.h>

// ---------------------------------------------------------------------------
// Problem constants (LeViT attention, fp32 reference; compute in bf16 WMMA
// with fp32 accumulation).
// ---------------------------------------------------------------------------
constexpr int BATCH = 256;
constexpr int NTOK  = 196;
constexpr int CDIM  = 512;
constexpr int HEADS = 8;
constexpr int KD    = 32;    // key dim
constexpr int DV    = 128;   // value dim per head
constexpr int DHD   = 1024;  // HEADS * DV
constexpr int HQKV  = 1536;  // HEADS * (KD + KD + DV)
constexpr int MROWS = BATCH * NTOK;       // 50176 (divisible by 128)
constexpr int NP    = 224;                // NTOK padded to 7*32
constexpr float SCALE = 0.17677669529663687f;  // 32^-0.5
constexpr float NEG_BIG = -1e30f;

// ---------------------------------------------------------------------------
// WMMA types & fragment loaders (CDNA5 16x16x32 bf16, wave32)
// ---------------------------------------------------------------------------
typedef __attribute__((ext_vector_type(16))) __bf16        bf16x16;
typedef __attribute__((ext_vector_type(8)))  float         f32x8;
typedef __attribute__((ext_vector_type(4)))  unsigned int  u32x4;

union Frag {
  bf16x16 v;
  u32x4   q[2];
};

__device__ __forceinline__ f32x8 f8zero() {
  f32x8 z;
#pragma unroll
  for (int i = 0; i < 8; ++i) z[i] = 0.0f;
  return z;
}

__device__ __forceinline__ unsigned short f2bf(float f) {
  unsigned int u = __float_as_uint(f);
  u += 0x7fffu + ((u >> 16) & 1u);   // round-to-nearest-even
  return (unsigned short)(u >> 16);
}

// A-matrix 16x32 (MxK) from row-major bf16 source, ld in elements.
// ISA layout: lanes 0-15 -> M=lane, K {0..7,16..23}; lanes 16-31 -> M=lane-16,
// K {8..15,24..31}; 2 consecutive K per VGPR. Two b128 loads per lane.
__device__ __forceinline__ Frag load_a_frag(const unsigned short* base, int ld, int lane) {
  const int r     = lane & 15;
  const int khalf = (lane >> 4) << 3;  // 0 or 8
  Frag f;
  f.q[0] = *reinterpret_cast<const u32x4*>(base + (size_t)r * ld + khalf);
  f.q[1] = *reinterpret_cast<const u32x4*>(base + (size_t)r * ld + 16 + khalf);
  return f;
}

// B-matrix 32x16 (KxN) built as W^T from row-major W[N,K] (ld in elements):
// lanes 0-15 -> N=lane, K {0..7,16..23}; lanes 16-31 -> N=lane-16, K {8..15,24..31}.
__device__ __forceinline__ Frag load_bt_frag(const unsigned short* w, int ld,
                                             int colbase, int kbase, int lane) {
  const int n    = lane & 15;
  const int koff = (lane >> 4) << 3;   // 0 or 8
  const unsigned short* src = w + (size_t)(colbase + n) * ld + kbase + koff;
  Frag f;
  f.q[0] = *reinterpret_cast<const u32x4*>(src);
  f.q[1] = *reinterpret_cast<const u32x4*>(src + 16);
  return f;
}

__device__ __forceinline__ f32x8 wmma_bf16(const Frag& a, const Frag& b, f32x8 c) {
  return __builtin_amdgcn_wmma_f32_16x16x32_bf16(false, a.v, false, b.v,
                                                 (short)0, c, false, false);
}

// ---------------------------------------------------------------------------
// CDNA5 async copy: 16B global -> LDS per lane, ASYNCcnt-tracked (VGLOBAL
// GV mode: VDST = LDS byte offset, VADDR = 64-bit global address).
// ---------------------------------------------------------------------------
__device__ __forceinline__ void async16(void* lds, const void* gptr) {
  unsigned int l = (unsigned int)(unsigned long long)(uintptr_t)lds;
  asm volatile("global_load_async_to_lds_b128 %0, %1, off"
               :: "v"(l), "v"(gptr)
               : "memory");
}

__device__ __forceinline__ void wait_asynccnt_le4() {
  asm volatile("s_wait_asynccnt 4" ::: "memory");
}
__device__ __forceinline__ void wait_asynccnt_0() {
  asm volatile("s_wait_asynccnt 0" ::: "memory");
}

// ---------------------------------------------------------------------------
// Kernel 1: fp32 -> bf16 conversion (vectorized x4)
// ---------------------------------------------------------------------------
__global__ void __launch_bounds__(256)
cvt_f32_bf16(const float* __restrict__ src, unsigned short* __restrict__ dst, int n4) {
  int i = blockIdx.x * 256 + threadIdx.x;
  if (i < n4) {
    float4 v = reinterpret_cast<const float4*>(src)[i];
    unsigned int lo = (unsigned int)f2bf(v.x) | ((unsigned int)f2bf(v.y) << 16);
    unsigned int hi = (unsigned int)f2bf(v.z) | ((unsigned int)f2bf(v.w) << 16);
    reinterpret_cast<uint2*>(dst)[i] = make_uint2(lo, hi);
  }
}

// ---------------------------------------------------------------------------
// Kernel 2: gather attention bias -> PADDED [HEADS, NP, NP] fp32.
// Padded cols/rows hold -1e30 so the softmax mask is baked in and the
// attention kernel's bias add is a single unconditional load+add.
// ---------------------------------------------------------------------------
__global__ void __launch_bounds__(256)
bias_gather(const float* __restrict__ ab, const int* __restrict__ idxs,
            float* __restrict__ biasp, int n_off) {
  int i = blockIdx.x * 256 + threadIdx.x;  // over HEADS*NP*NP
  if (i < HEADS * NP * NP) {
    const int h   = i / (NP * NP);
    const int rem = i % (NP * NP);
    const int m   = rem / NP;
    const int c   = rem % NP;
    float v = NEG_BIG;
    if (m < NTOK && c < NTOK) v = ab[h * n_off + idxs[m * NTOK + c]];
    biasp[i] = v;
  }
}

// ---------------------------------------------------------------------------
// Tiled GEMM with async-LDS double buffering.
//   C[M, NCOLS] = A[M, KDIM] @ W[NCOLS, KDIM]^T + bias
// Block tile 128x128, K-step 32. 256 threads = 8 waves in a 2(M) x 4(N) grid,
// each wave computing 64x32 (4x2 WMMA frags). Per K-step, 16 KB (A+B tiles)
// is staged with global_load_async_to_lds_b128 (4 issues/thread), waited with
// s_wait_asynccnt, then consumed via ds_load_b128 fragment loads.
// Epilogue: QKV_EPI scatters into padded q/k/vT bf16 buffers; else fp32 out.
// ---------------------------------------------------------------------------
template <int NCOLS, int KDIM, bool QKV_EPI>
__global__ void __launch_bounds__(256)
gemm_tiled(const unsigned short* __restrict__ A,   // [M, KDIM]
           const unsigned short* __restrict__ W,   // [NCOLS, KDIM]
           const float* __restrict__ bias,         // [NCOLS]
           unsigned short* __restrict__ qbuf,
           unsigned short* __restrict__ kbuf,
           unsigned short* __restrict__ vtbuf,
           float* __restrict__ out) {
  __shared__ unsigned short sA[2][128][32];   // 16 KB
  __shared__ unsigned short sB[2][128][32];   // 16 KB

  const int tid  = threadIdx.x;
  const int lane = tid & 31;
  const int wid  = tid >> 5;
  const int wm   = wid >> 2;      // 0..1  (64-row slice)
  const int wn   = wid & 3;       // 0..3  (32-col slice)

  const int nbn    = NCOLS / 128;
  const int rowblk = (blockIdx.x / nbn) * 128;
  const int colblk = (blockIdx.x % nbn) * 128;

  const unsigned short* gA = A + (size_t)rowblk * KDIM;
  const unsigned short* gW = W + (size_t)colblk * KDIM;
  const int r = tid >> 2;            // 0..63
  const int c = (tid & 3) * 8;       // element offset: 0,8,16,24 (16B chunks)

  auto stage = [&](int bufi, int kk) {
    async16(&sA[bufi][r][c],      gA + (size_t)r        * KDIM + kk + c);
    async16(&sA[bufi][r + 64][c], gA + (size_t)(r + 64) * KDIM + kk + c);
    async16(&sB[bufi][r][c],      gW + (size_t)r        * KDIM + kk + c);
    async16(&sB[bufi][r + 64][c], gW + (size_t)(r + 64) * KDIM + kk + c);
  };

  f32x8 acc[4][2];
#pragma unroll
  for (int i = 0; i < 4; ++i)
#pragma unroll
    for (int j = 0; j < 2; ++j) acc[i][j] = f8zero();

  constexpr int nK = KDIM / 32;
  stage(0, 0);
  int buf = 0;
#pragma unroll 1
  for (int s = 0; s < nK; ++s) {
    if (s + 1 < nK) {
      stage(buf ^ 1, (s + 1) * 32);
      wait_asynccnt_le4();        // drain the 4 issues of stage s
    } else {
      wait_asynccnt_0();
    }
    __syncthreads();              // all waves' staging of buf complete

    const unsigned short* Ab = &sA[buf][wm * 64][0];
    const unsigned short* Bb = &sB[buf][wn * 32][0];
    Frag af[4], bf2[2];
#pragma unroll
    for (int i = 0; i < 4; ++i) af[i] = load_a_frag(Ab + i * 16 * 32, 32, lane);
#pragma unroll
    for (int j = 0; j < 2; ++j) bf2[j] = load_bt_frag(Bb, 32, j * 16, 0, lane);
#pragma unroll
    for (int i = 0; i < 4; ++i)
#pragma unroll
      for (int j = 0; j < 2; ++j)
        acc[i][j] = wmma_bf16(af[i], bf2[j], acc[i][j]);

    __syncthreads();              // done reading buf before it is re-staged
    buf ^= 1;
  }

  // D layout: lane n = lane&15; VGPR v -> m = v + 8*(lane>>4)
  const int n  = lane & 15;
  const int mb = (lane >> 4) << 3;
#pragma unroll
  for (int i = 0; i < 4; ++i) {
#pragma unroll
    for (int j = 0; j < 2; ++j) {
      const int col = colblk + wn * 32 + j * 16 + n;
      const float bj = bias[col];
      if (QKV_EPI) {
        const int h = col / (2 * KD + DV);      // /192
        const int rr = col % (2 * KD + DV);
#pragma unroll
        for (int v = 0; v < 8; ++v) {
          const int row = rowblk + wm * 64 + i * 16 + mb + v;
          const int b   = row / NTOK;
          const int tok = row % NTOK;
          const size_t bh = (size_t)(b * HEADS + h);
          const unsigned short bv = f2bf(acc[i][j][v] + bj);
          if (rr < KD)            qbuf[(bh * NP + tok) * KD + rr] = bv;
          else if (rr < 2 * KD)   kbuf[(bh * NP + tok) * KD + (rr - KD)] = bv;
          else                    vtbuf[(bh * DV + (rr - 2 * KD)) * NP + tok] = bv;
        }
      } else {
#pragma unroll
        for (int v = 0; v < 8; ++v) {
          const int row = rowblk + wm * 64 + i * 16 + mb + v;
          out[(size_t)row * NCOLS + col] = acc[i][j][v] + bj;
        }
      }
    }
  }
}

// ---------------------------------------------------------------------------
// Fused attention per (b,h). 128 threads = 4 waves; each wave owns 16-row
// tiles (rt = wave, wave+4, ...). S tile [16 x NP] fp32 lives in the wave's
// private LDS slab (no cross-wave sharing -> no barriers; same-wave DS ops
// are in-order). Softmax uses 2 lanes/row + __shfl_xor(16) (wave32), with
// 4-way strided unrolling (112 = 4*28 exact) and independent accumulators to
// batch ds_loads and break the serial max/sum chains.
// P is rewritten in place as bf16, then PV runs 7 K-steps x 8 WMMA accums.
// ---------------------------------------------------------------------------
__global__ void __launch_bounds__(128)
attn_kernel(const unsigned short* __restrict__ qb,
            const unsigned short* __restrict__ kb,
            const unsigned short* __restrict__ vtb,
            const float* __restrict__ biasp,     // [HEADS, NP, NP], mask baked in
            unsigned short* __restrict__ oh) {   // [MROWS, DHD] bf16
  __shared__ float sS[4][16][NP];                // 57,344 bytes
  const int lane = threadIdx.x & 31;
  const int wv   = threadIdx.x >> 5;
  const int bh   = blockIdx.x;
  const int b    = bh >> 3;
  const int h    = bh & 7;
  const unsigned short* qp = qb  + (size_t)bh * NP * KD;
  const unsigned short* kp = kb  + (size_t)bh * NP * KD;
  const unsigned short* vp = vtb + (size_t)bh * DV * NP;
  const float* bp = biasp + (size_t)h * NP * NP;

  const int n  = lane & 15;
  const int mb = (lane >> 4) << 3;

#pragma unroll 1
  for (int rt = wv; rt < 13; rt += 4) {          // 13 row tiles cover 196
    const int row0 = rt * 16;
    Frag qa = load_a_frag(qp + (size_t)row0 * KD, KD, lane);  // K = KD = 32

    // ---- S = q k^T * scale + biasp  -> LDS (unconditional, mask in bias) --
#pragma unroll 2
    for (int ct = 0; ct < NP / 16; ++ct) {
      Frag kf = load_bt_frag(kp, KD, ct * 16, 0, lane);
      f32x8 s = wmma_bf16(qa, kf, f8zero());
      const int col = ct * 16 + n;
      const float* bcol = bp + (size_t)(row0 + mb) * NP + col;
#pragma unroll
      for (int v = 0; v < 8; ++v)
        sS[wv][mb + v][col] = s[v] * SCALE + bcol[v * NP];
    }

    // ---- softmax over NP cols, 2 lanes per row, 4-way unrolled ----------
    {
      const int m  = lane & 15;
      const int hf = lane >> 4;
      float* row = &sS[wv][m][0];

      float mx0 = NEG_BIG, mx1 = NEG_BIG, mx2 = NEG_BIG, mx3 = NEG_BIG;
#pragma unroll 1
      for (int cc = hf; cc < NP; cc += 8) {
        mx0 = fmaxf(mx0, row[cc]);
        mx1 = fmaxf(mx1, row[cc + 2]);
        mx2 = fmaxf(mx2, row[cc + 4]);
        mx3 = fmaxf(mx3, row[cc + 6]);
      }
      float mx = fmaxf(fmaxf(mx0, mx1), fmaxf(mx2, mx3));
      mx = fmaxf(mx, __shfl_xor(mx, 16, 32));

      float s0 = 0.0f, s1 = 0.0f, s2 = 0.0f, s3 = 0.0f;
#pragma unroll 1
      for (int cc = hf; cc < NP; cc += 8) {
        float e0 = __expf(row[cc]     - mx);
        float e1 = __expf(row[cc + 2] - mx);
        float e2 = __expf(row[cc + 4] - mx);
        float e3 = __expf(row[cc + 6] - mx);
        row[cc] = e0; row[cc + 2] = e1; row[cc + 4] = e2; row[cc + 6] = e3;
        s0 += e0; s1 += e1; s2 += e2; s3 += e3;
      }
      float sum = (s0 + s1) + (s2 + s3);
      sum += __shfl_xor(sum, 16, 32);
      const float inv = 1.0f / sum;

      // in-place fp32 -> bf16: a ushort write at index j only overlaps the
      // float at index j/2, which both lane-halves have already read.
      unsigned short* prow = reinterpret_cast<unsigned short*>(row);
#pragma unroll 1
      for (int cc = hf; cc < NP; cc += 8) {
        unsigned short p0 = f2bf(row[cc]     * inv);
        unsigned short p1 = f2bf(row[cc + 2] * inv);
        unsigned short p2 = f2bf(row[cc + 4] * inv);
        unsigned short p3 = f2bf(row[cc + 6] * inv);
        prow[cc] = p0; prow[cc + 2] = p1; prow[cc + 4] = p2; prow[cc + 6] = p3;
      }
    }

    // ---- out = P @ V  (P bf16 in LDS, row stride 2*NP ushorts; V^T global)
    f32x8 acc[8];
#pragma unroll
    for (int nt = 0; nt < 8; ++nt) acc[nt] = f8zero();
    const unsigned short* pbase = reinterpret_cast<const unsigned short*>(&sS[wv][0][0]);
#pragma unroll 1
    for (int kt = 0; kt < NP / 32; ++kt) {
      Frag pa = load_a_frag(pbase + kt * 32, 2 * NP, lane);
#pragma unroll
      for (int nt = 0; nt < 8; ++nt) {
        Frag vf = load_bt_frag(vp, NP, nt * 16, kt * 32, lane);
        acc[nt] = wmma_bf16(pa, vf, acc[nt]);
      }
    }
#pragma unroll
    for (int nt = 0; nt < 8; ++nt) {
#pragma unroll
      for (int v = 0; v < 8; ++v) {
        const int tok = row0 + mb + v;
        if (tok < NTOK)
          oh[((size_t)(b * NTOK + tok)) * DHD + h * DV + nt * 16 + n] =
              f2bf(acc[nt][v]);
      }
    }
  }
}

// ---------------------------------------------------------------------------
// Host launcher
// ---------------------------------------------------------------------------
extern "C" void kernel_launch(void* const* d_in, const int* in_sizes, int n_in,
                              void* d_out, int out_size, void* d_ws, size_t ws_size,
                              hipStream_t stream) {
  const float* x      = (const float*)d_in[0];
  const float* qkv_w  = (const float*)d_in[1];
  const float* qkv_b  = (const float*)d_in[2];
  const float* proj_w = (const float*)d_in[3];
  const float* proj_b = (const float*)d_in[4];
  const float* ab     = (const float*)d_in[5];
  const int*   idxs   = (const int*)d_in[6];
  float* out = (float*)d_out;
  const int n_off = in_sizes[5] / HEADS;

  // Workspace carve (256B aligned)
  char* p = (char*)d_ws;
  auto carve = [&](size_t bytes) -> void* {
    void* r = (void*)p;
    p += (bytes + 255) & ~(size_t)255;
    return r;
  };
  unsigned short* x_bf  = (unsigned short*)carve((size_t)MROWS * CDIM * 2);
  unsigned short* wq_bf = (unsigned short*)carve((size_t)HQKV * CDIM * 2);
  unsigned short* wp_bf = (unsigned short*)carve((size_t)CDIM * DHD * 2);
  unsigned short* qbuf  = (unsigned short*)carve((size_t)BATCH * HEADS * NP * KD * 2);
  unsigned short* kbuf  = (unsigned short*)carve((size_t)BATCH * HEADS * NP * KD * 2);
  unsigned short* vtbuf = (unsigned short*)carve((size_t)BATCH * HEADS * DV * NP * 2);
  float*          biasp = (float*)carve((size_t)HEADS * NP * NP * 4);
  unsigned short* ohbuf = (unsigned short*)carve((size_t)MROWS * DHD * 2);
  (void)ws_size; (void)n_in; (void)out_size;

  // Zero the padded q/k/vT buffers (padding rows/cols must be 0)
  hipMemsetAsync(qbuf,  0, (size_t)BATCH * HEADS * NP * KD * 2, stream);
  hipMemsetAsync(kbuf,  0, (size_t)BATCH * HEADS * NP * KD * 2, stream);
  hipMemsetAsync(vtbuf, 0, (size_t)BATCH * HEADS * DV * NP * 2, stream);

  // fp32 -> bf16 conversions
  {
    int n4 = MROWS * CDIM / 4;
    cvt_f32_bf16<<<(n4 + 255) / 256, 256, 0, stream>>>(x, x_bf, n4);
  }
  {
    int n4 = HQKV * CDIM / 4;
    cvt_f32_bf16<<<(n4 + 255) / 256, 256, 0, stream>>>(qkv_w, wq_bf, n4);
  }
  {
    int n4 = CDIM * DHD / 4;
    cvt_f32_bf16<<<(n4 + 255) / 256, 256, 0, stream>>>(proj_w, wp_bf, n4);
  }

  // Padded bias gather (mask baked into padding)
  bias_gather<<<(HEADS * NP * NP + 255) / 256, 256, 0, stream>>>(ab, idxs, biasp, n_off);

  // QKV GEMM: 128x128 block tiles, async-LDS double buffered
  {
    int grid = (MROWS / 128) * (HQKV / 128);   // 392 * 12 = 4704
    gemm_tiled<HQKV, CDIM, true><<<grid, 256, 0, stream>>>(
        x_bf, wq_bf, qkv_b, qbuf, kbuf, vtbuf, nullptr);
  }

  // Fused attention: one block per (b,h)
  attn_kernel<<<BATCH * HEADS, 128, 0, stream>>>(qbuf, kbuf, vtbuf, biasp, ohbuf);

  // Projection GEMM
  {
    int grid = (MROWS / 128) * (CDIM / 128);   // 392 * 4 = 1568
    gemm_tiled<CDIM, DHD, false><<<grid, 256, 0, stream>>>(
        ohbuf, wp_bf, proj_b, nullptr, nullptr, nullptr, out);
  }
}
